// Sampler_15049565405579
// MI455X (gfx1250) — compile-verified, matrix operand order checked by reading, not run on hardware
//
#include <hip/hip_runtime.h>
#include <hip/hip_bf16.h>
#include <cstdint>

#define TPB        256      // 8 wave32 per block
#define NBINS      1024
#define BIN_SCALE  32.0f    // bins per unit of z-M; covered range = NBINS/BIN_SCALE = 32
#define HASH_SLOTS 512
#define HASH_MASK  (HASH_SLOTS - 1)
#define SAMP_EPS   1e-5f

// ---------------- CDNA5 async-copy helpers (gfx1250) ----------------
// Per-lane 16B global -> LDS, tracked by ASYNCcnt (CDNA5 ISA).
__device__ __forceinline__ void async_load_b128(uint32_t lds_addr, const void* gptr) {
  asm volatile("global_load_async_to_lds_b128 %0, %1, off"
               :: "v"(lds_addr), "v"((unsigned long long)(uintptr_t)gptr)
               : "memory");
}
__device__ __forceinline__ void wait_async(int n) {
  if (n == 0) asm volatile("s_wait_asynccnt 0x0" ::: "memory");
  else        asm volatile("s_wait_asynccnt 0x1" ::: "memory");
}

// ---------------- WMMA wave reduction ----------------
// A = ones, B = per-lane partial replicated. C[m,n] = sum_k B[k,n], so
// sum(all 256 C elems) = 16 * sum(all B elems) = 16*elemsPerLane*sum(partials).
// Layout-independent: just sum every lane's 8 accumulator floats.
__device__ __forceinline__ float wave_sum32(float x) {
  typedef float v8f __attribute__((ext_vector_type(8)));
#if __has_builtin(__builtin_amdgcn_wmma_f32_16x16x4_f32)
  typedef float v2f __attribute__((ext_vector_type(2)));
  v2f a; a[0] = 1.0f; a[1] = 1.0f;
  v2f b; b[0] = x;    b[1] = x;
  v8f c = {};
  v8f d = __builtin_amdgcn_wmma_f32_16x16x4_f32(false, a, false, b, (short)0, c, false, false);
  float s = d[0]+d[1]+d[2]+d[3]+d[4]+d[5]+d[6]+d[7];
  #pragma unroll
  for (int off = 16; off > 0; off >>= 1) s += __shfl_xor(s, off, 32);
  return s * (1.0f / 32.0f);          // grand total = 16*2*sum(partials)
#else
  typedef _Float16 v16h __attribute__((ext_vector_type(16)));
  v16h a, b;
  _Float16 hx = (_Float16)x;
  #pragma unroll
  for (int i = 0; i < 16; ++i) { a[i] = (_Float16)1.0f; b[i] = hx; }
  v8f c = {};
  v8f d = __builtin_amdgcn_wmma_f32_16x16x32_f16(false, a, false, b, (short)0, c, false, false);
  float s = d[0]+d[1]+d[2]+d[3]+d[4]+d[5]+d[6]+d[7];
  #pragma unroll
  for (int off = 16; off > 0; off >>= 1) s += __shfl_xor(s, off, 32);
  return s * (1.0f / 256.0f);         // grand total = 16*16*sum(partials)
#endif
}

// ---------------- penalty hash table in LDS ----------------
__device__ __forceinline__ uint32_t hash_tok(int t) {
  return ((uint32_t)t * 2654435761u >> 23) & HASH_MASK;
}

__device__ void build_hash(int* h_tok, float* h_val, const int* toks_row, int H,
                           float pres, float freq, bool active) {
  int tid = threadIdx.x;
  for (int s = tid; s < HASH_SLOTS; s += TPB) { h_tok[s] = -1; h_val[s] = 0.0f; }
  __syncthreads();
  if (active) {
    for (int j = tid; j < H; j += TPB) {
      int t = toks_row[j];
      uint32_t s = hash_tok(t);
      while (true) {
        int old = atomicCAS(&h_tok[s], -1, t);
        if (old == -1 || old == t) { atomicAdd(&h_val[s], 1.0f); break; }
        s = (s + 1) & HASH_MASK;
      }
    }
  }
  __syncthreads();
  for (int s = tid; s < HASH_SLOTS; s += TPB)
    if (h_tok[s] != -1) h_val[s] = freq * h_val[s] + pres;
  __syncthreads();
}

__device__ __forceinline__ float pen_lookup(const int* h_tok, const float* h_val, int i) {
  uint32_t s = hash_tok(i);
  while (true) {
    int t = h_tok[s];
    if (t == i)  return h_val[s];
    if (t == -1) return 0.0f;
    s = (s + 1) & HASH_MASK;
  }
}

__device__ __forceinline__ int bin_of(float t) {   // t = z - M, t <= 0
  int b = (int)(-t * BIN_SCALE);
  b = (NBINS - 1) - min(b, NBINS - 1);
  return b;
}

// ============ kernel 1: row max of raw logits, scaled by 1/T ============
// Penalties only lower values and touch <=200 of V entries; using the
// unpenalized max as the softmax shift M is exact for the output (softmax is
// shift-invariant when the same M is used for exp and Z) and keeps z-M <= 0.
__global__ void __launch_bounds__(TPB)
k1_rowmax(const float* __restrict__ logits, const float* temp_v,
          float* __restrict__ Mrow, int V) {
  __shared__ float4 stage[2][TPB];
  __shared__ float  red[TPB / 32];

  int row = blockIdx.x, tid = threadIdx.x;
  const float* lrow = logits + (size_t)row * V;
  int ntiles = V / (TPB * 4);
  float m = -3.0e38f;
  uint32_t lds0 = (uint32_t)(uintptr_t)&stage[0][tid];
  uint32_t lds1 = (uint32_t)(uintptr_t)&stage[1][tid];

  if (ntiles > 0) {
    async_load_b128(lds0, lrow + tid * 4);
    for (int it = 0; it < ntiles; ++it) {
      if (it + 1 < ntiles)
        async_load_b128((it & 1) ? lds0 : lds1,
                        lrow + (size_t)(it + 1) * (TPB * 4) + tid * 4);
      wait_async((it + 1 < ntiles) ? 1 : 0);
      float4 v = stage[it & 1][tid];
      m = fmaxf(m, fmaxf(fmaxf(v.x, v.y), fmaxf(v.z, v.w)));
    }
  }
  for (int i = ntiles * (TPB * 4) + tid; i < V; i += TPB) m = fmaxf(m, lrow[i]);

  #pragma unroll
  for (int off = 16; off > 0; off >>= 1) m = fmaxf(m, __shfl_xor(m, off, 32));
  if ((tid & 31) == 0) red[tid >> 5] = m;
  __syncthreads();
  if (tid == 0) {
    float mm = red[0];
    for (int w = 1; w < TPB / 32; ++w) mm = fmaxf(mm, red[w]);
    Mrow[row] = mm * (1.0f / temp_v[row]);   // M in z-units (invT > 0, monotone)
  }
}

// ============ kernel 2: exp-sum Z, histogram over z-M, cutoff scan ============
__global__ void __launch_bounds__(TPB)
k2_stats(const float* __restrict__ logits, const int* __restrict__ toks,
         const float* pres_v, const float* freq_v, const float* temp_v,
         const float* __restrict__ Mrow, const float* topp_v, const int* topk_v,
         float* __restrict__ Zrow, int* __restrict__ binB, int* __restrict__ budget,
         int V, int H) {
  __shared__ int      h_tok[HASH_SLOTS];
  __shared__ float    h_val[HASH_SLOTS];
  __shared__ float4   stage[2][TPB];
  __shared__ uint32_t hcnt[NBINS];
  __shared__ float    hsum[NBINS];
  __shared__ float    red[TPB / 32];

  int row = blockIdx.x, tid = threadIdx.x;
  for (int b = tid; b < NBINS; b += TPB) { hcnt[b] = 0u; hsum[b] = 0.0f; }
  float pres = pres_v[row], freq = freq_v[row];
  bool active = (pres >= SAMP_EPS) || (freq >= SAMP_EPS);
  float invT = 1.0f / temp_v[row];
  build_hash(h_tok, h_val, toks + (size_t)row * H, H, pres, freq, active); // has barriers

  float M = Mrow[row];
  const float* lrow = logits + (size_t)row * V;
  int ntiles = V / (TPB * 4);
  float s = 0.0f;
  uint32_t lds0 = (uint32_t)(uintptr_t)&stage[0][tid];
  uint32_t lds1 = (uint32_t)(uintptr_t)&stage[1][tid];

  if (ntiles > 0) {
    async_load_b128(lds0, lrow + tid * 4);
    for (int it = 0; it < ntiles; ++it) {
      if (it + 1 < ntiles)
        async_load_b128((it & 1) ? lds0 : lds1,
                        lrow + (size_t)(it + 1) * (TPB * 4) + tid * 4);
      wait_async((it + 1 < ntiles) ? 1 : 0);
      float4 v = stage[it & 1][tid];
      int base = it * (TPB * 4) + tid * 4;
      float e4[4] = {v.x, v.y, v.z, v.w};
      #pragma unroll
      for (int j = 0; j < 4; ++j) {
        float p = active ? pen_lookup(h_tok, h_val, base + j) : 0.0f;
        float z = (e4[j] - p) * invT;
        float e = __expf(z - M);
        s += e;
        int b = bin_of(z - M);
        atomicAdd(&hcnt[b], 1u);
        atomicAdd(&hsum[b], e);
      }
    }
  }
  for (int i = ntiles * (TPB * 4) + tid; i < V; i += TPB) {
    float p = active ? pen_lookup(h_tok, h_val, i) : 0.0f;
    float z = (lrow[i] - p) * invT;
    float e = __expf(z - M);
    s += e;
    int b = bin_of(z - M);
    atomicAdd(&hcnt[b], 1u);
    atomicAdd(&hsum[b], e);
  }

  float wsum = wave_sum32(s);                 // WMMA cross-lane reduction (EXEC full)
  if ((tid & 31) == 0) red[tid >> 5] = wsum;
  __syncthreads();                            // also makes hcnt/hsum atomics visible

  if (tid == 0) {
    float Z = 0.0f;
    for (int w = 0; w < TPB / 32; ++w) Z += red[w];
    Zrow[row] = Z;
    // ---- cutoff scan: walk bins high->low until top_p or top_k is crossed ----
    float invZ = 1.0f / Z;
    float topp = topp_v[row];
    int   topk = topk_v[row];
    float cp = 0.0f; int cc = 0, B = -1, kB = 0;
    for (int b = NBINS - 1; b >= 0; --b) {
      uint32_t c = hcnt[b];
      if (!c) continue;
      float p = hsum[b] * invZ;
      if ((cc + (int)c <= topk) && (cp + p <= topp)) { cc += (int)c; cp += p; continue; }
      B = b;
      float avg = p / (float)c;
      int byK = topk - cc;
      int byP = (cp > topp) ? 0 : (int)((topp - cp) / fmaxf(avg, 1e-30f)) + 1;
      kB = min(byK, byP); kB = min(kB, (int)c); if (kB < 0) kB = 0;
      break;
    }
    if (cc == 0 && kB < 1) kB = 1;   // reference always keeps the argmax
    binB[row]   = B;
    budget[row] = kB;
  }
}

// ============ kernel 3: write filtered probabilities ============
__global__ void __launch_bounds__(TPB)
k3_write(const float* __restrict__ logits, const int* __restrict__ toks,
         const float* pres_v, const float* freq_v, const float* temp_v,
         const float* __restrict__ Mrow, const float* __restrict__ Zrow,
         const int* __restrict__ binB, int* __restrict__ budget,
         float* __restrict__ out, int V, int H) {
  __shared__ int    h_tok[HASH_SLOTS];
  __shared__ float  h_val[HASH_SLOTS];
  __shared__ float4 stage[2][TPB];

  int row = blockIdx.x, tid = threadIdx.x;
  float pres = pres_v[row], freq = freq_v[row];
  bool active = (pres >= SAMP_EPS) || (freq >= SAMP_EPS);
  float invT = 1.0f / temp_v[row];
  build_hash(h_tok, h_val, toks + (size_t)row * H, H, pres, freq, active);

  float M = Mrow[row];
  float invZ = 1.0f / Zrow[row];
  int   B = binB[row];
  const float* lrow = logits + (size_t)row * V;
  float* orow = out + (size_t)row * V;
  int ntiles = V / (TPB * 4);
  uint32_t lds0 = (uint32_t)(uintptr_t)&stage[0][tid];
  uint32_t lds1 = (uint32_t)(uintptr_t)&stage[1][tid];

  if (ntiles > 0) {
    async_load_b128(lds0, lrow + tid * 4);
    for (int it = 0; it < ntiles; ++it) {
      if (it + 1 < ntiles)
        async_load_b128((it & 1) ? lds0 : lds1,
                        lrow + (size_t)(it + 1) * (TPB * 4) + tid * 4);
      wait_async((it + 1 < ntiles) ? 1 : 0);
      float4 v = stage[it & 1][tid];
      int base = it * (TPB * 4) + tid * 4;
      float e4[4] = {v.x, v.y, v.z, v.w};
      float o4[4];
      #pragma unroll
      for (int j = 0; j < 4; ++j) {
        float pn = active ? pen_lookup(h_tok, h_val, base + j) : 0.0f;
        float z  = (e4[j] - pn) * invT;
        float p  = __expf(z - M) * invZ;
        int   b  = bin_of(z - M);
        float o  = 0.0f;
        if (b > B) o = p;
        else if (b == B) { int old = atomicAdd(&budget[row], -1); if (old > 0) o = p; }
        o4[j] = o;
      }
      float4 ov = {o4[0], o4[1], o4[2], o4[3]};
      *(float4*)(orow + base) = ov;            // coalesced global_store_b128
    }
  }
  for (int i = ntiles * (TPB * 4) + tid; i < V; i += TPB) {
    float pn = active ? pen_lookup(h_tok, h_val, i) : 0.0f;
    float z  = (lrow[i] - pn) * invT;
    float p  = __expf(z - M) * invZ;
    int   b  = bin_of(z - M);
    float o  = 0.0f;
    if (b > B) o = p;
    else if (b == B) { int old = atomicAdd(&budget[row], -1); if (old > 0) o = p; }
    orow[i] = o;
  }
}

extern "C" void kernel_launch(void* const* d_in, const int* in_sizes, int n_in,
                              void* d_out, int out_size, void* d_ws, size_t ws_size,
                              hipStream_t stream) {
  const float* logits = (const float*)d_in[0];
  const float* pres   = (const float*)d_in[1];
  const float* freq   = (const float*)d_in[2];
  const float* temp   = (const float*)d_in[3];
  const float* topp   = (const float*)d_in[4];
  const int*   toks   = (const int*)d_in[5];
  const int*   topk   = (const int*)d_in[6];

  int N = in_sizes[1];
  int V = in_sizes[0] / N;
  int H = in_sizes[5] / N;

  char* ws = (char*)d_ws;
  float* Mrow   = (float*)ws;  ws += (size_t)N * sizeof(float);
  float* Zrow   = (float*)ws;  ws += (size_t)N * sizeof(float);
  int*   binB   = (int*)ws;    ws += (size_t)N * sizeof(int);
  int*   budget = (int*)ws;
  float* out    = (float*)d_out;

  k1_rowmax<<<N, TPB, 0, stream>>>(logits, temp, Mrow, V);
  k2_stats <<<N, TPB, 0, stream>>>(logits, toks, pres, freq, temp, Mrow,
                                   topp, topk, Zrow, binB, budget, V, H);
  k3_write <<<N, TPB, 0, stream>>>(logits, toks, pres, freq, temp, Mrow, Zrow,
                                   binB, budget, out, V, H);
}
